// MiniPointNet_16449724744454
// MI455X (gfx1250) — matrix operand order repaired and checked by reading, not compiled, not run on hardware
//
#include <hip/hip_runtime.h>

typedef __attribute__((ext_vector_type(16))) _Float16 v16h;
typedef __attribute__((ext_vector_type(8)))  _Float16 v8h;
typedef __attribute__((ext_vector_type(8)))  float    v8f;

static constexpr int    kN        = 1048576;          // total points
static constexpr int    kC        = 128;              // feature dim
static constexpr int    kSegs     = 4096;
static constexpr int    kRowsBlk  = 128;              // rows per block (8 waves x 16 rows)
static constexpr int    kGridMain = kN / kRowsBlk;    // 8192
static constexpr float  kInvN     = 1.0f / 1048576.0f;
static constexpr float  kEps      = 1e-5f;

// ---- workspace layout (bytes) ----
static constexpr size_t Y_OFF     = 0;                                       // N*128 f16  (256 MB)
static constexpr size_t PART_OFF  = (size_t)kN * kC * 2;                     // 8192*256 f32 (8 MB)
static constexpr size_t PART2_OFF = PART_OFF + (size_t)kGridMain * 256 * 4;  // 256*256 f32
static constexpr size_t STATS_OFF = PART2_OFF + (size_t)256 * 256 * 4;       // 256 f32
static constexpr size_t WS_NEED   = STATS_OFF + 256 * 4;

__device__ __forceinline__ float mish_f(float x) {
  // x * tanh(softplus(x)) == x * g/(g+2), g = e^x (e^x + 2); clamp avoids inf/inf
  float e = __expf(fminf(x, 15.0f));
  float g = e * (e + 2.0f);
  return x * (g / (g + 2.0f));
}

// ---------------- pass 0: points @ w_first + b_first -> mish -> stats -> y (f16) ----------------
__global__ void __launch_bounds__(256) k_first(const float* __restrict__ pts,
                                               const float* __restrict__ wf,
                                               const float* __restrict__ bf,
                                               _Float16* __restrict__ y,
                                               float* __restrict__ part) {
  __shared__ float lpts[256];            // 128 rows x 2 coords
  __shared__ float lsum[256];
  __shared__ float lsq[256];
  const int t = threadIdx.x;
  const long long row0 = (long long)blockIdx.x * kRowsBlk;
  lpts[t] = pts[row0 * 2 + t];
  __syncthreads();
  const int c = t & 127, h = t >> 7;
  const float w0 = wf[c], w1 = wf[kC + c], bb = bf[c];
  float s = 0.0f, q = 0.0f;
#pragma unroll 4
  for (int j = 0; j < 64; ++j) {
    const int r = h * 64 + j;
    float x = fmaf(lpts[2 * r], w0, fmaf(lpts[2 * r + 1], w1, bb));
    float v = mish_f(x);
    s += v; q += v * v;
    y[(row0 + r) * kC + c] = (_Float16)v;
  }
  lsum[t] = s; lsq[t] = q;
  __syncthreads();
  if (t < 128) {
    const long long pb = (long long)blockIdx.x * 256;
    part[pb + t]       = lsum[t] + lsum[t + 128];
    part[pb + 128 + t] = lsq[t]  + lsq[t + 128];
  }
}

// ---------------- reduction stage 1: 8192 block-partials -> 256 ----------------
__global__ void __launch_bounds__(256) k_reduce1(const float* __restrict__ part,
                                                 float* __restrict__ part2) {
  const int t = threadIdx.x;
  const long long b = blockIdx.x;      // 256 blocks
  float acc = 0.0f;
#pragma unroll 4
  for (int i = 0; i < 32; ++i) acc += part[(b * 32 + i) * 256 + t];
  part2[b * 256 + t] = acc;
}

// ---------------- reduction stage 2: fold stats into scale/shift ----------------
__global__ void __launch_bounds__(256) k_finalize(const float* __restrict__ part2,
                                                  const float* __restrict__ gamma,
                                                  const float* __restrict__ beta,
                                                  float* __restrict__ stats) {
  __shared__ float l[256];
  const int t = threadIdx.x;
  float acc = 0.0f;
  for (int i = 0; i < 256; ++i) acc += part2[i * 256 + t];   // fixed order: deterministic
  l[t] = acc;
  __syncthreads();
  if (t < 128) {
    float mean = l[t] * kInvN;
    float var  = l[t + 128] * kInvN - mean * mean;
    float rs   = rsqrtf(var + kEps);
    float sc   = gamma[t] * rs;
    stats[t]       = sc;                 // scale
    stats[128 + t] = beta[t] - mean * sc; // shift
  }
}

// ---------------- mid pass: normalize(y) -> WMMA matmul W(128x128)+b -> [mish] -> stats -> y ----
__global__ void __launch_bounds__(256) k_mid(_Float16* y,
                                             const float* __restrict__ W,
                                             const float* __restrict__ bvec,
                                             const float* __restrict__ stats,
                                             float* __restrict__ part,
                                             int do_mish) {
  __shared__ __attribute__((aligned(16))) _Float16 Wt[128 * 136];  // W transposed: Wt[n][k]
  __shared__ float lstats[256];
  __shared__ float lbias[128];
  __shared__ float lsum[8 * 128];
  __shared__ float lsq[8 * 128];

  const int t = threadIdx.x;
  // load W (f32, row-major [k][n]) -> LDS f16 transposed [n][k], padded stride 136
#pragma unroll 4
  for (int i = 0; i < 64; ++i) {
    const int idx = i * 256 + t;
    const int k = idx >> 7, n = idx & 127;
    Wt[n * 136 + k] = (_Float16)W[idx];
  }
  lstats[t] = stats[t];
  if (t < 128) lbias[t] = bvec[t];
  __syncthreads();

  const int wave = t >> 5, lane = t & 31;
  const int m = lane & 15, hh = lane >> 4;
  const long long row0 = (long long)blockIdx.x * kRowsBlk + wave * 16;

  // ---- A fragments: 16x32 f16 per k-chunk, normalized on the fly ----
  v16h A[4];
#pragma unroll
  for (int kc = 0; kc < 4; ++kc) {
    const int kb = kc * 32 + hh * 8;
    const _Float16* rowp = y + (row0 + m) * kC;
    v8h r0 = *(const v8h*)(rowp + kb);        // K = kb .. kb+7
    v8h r1 = *(const v8h*)(rowp + kb + 16);   // K = kb+16 .. kb+23
#pragma unroll
    for (int j = 0; j < 8; ++j) {
      float z0 = fmaf((float)r0[j], lstats[kb + j],      lstats[128 + kb + j]);
      float z1 = fmaf((float)r1[j], lstats[kb + 16 + j], lstats[128 + kb + 16 + j]);
      A[kc][j]     = (_Float16)z0;
      A[kc][j + 8] = (_Float16)z1;
    }
  }

  // ---- accumulate 16x128 output stripe with 32 WMMAs ----
  v8f D[8];
#pragma unroll
  for (int i = 0; i < 8; ++i) D[i] = (v8f)(0.0f);

#pragma unroll
  for (int kc = 0; kc < 4; ++kc) {
    const int kb2 = kc * 32 + hh * 8;      // B: half-wave h holds K 8h..8h+7 and 16+8h..16+8h+7
#pragma unroll
    for (int tn = 0; tn < 8; ++tn) {
      const int n = tn * 16 + (lane & 15);
      const v8h* bp = (const v8h*)(&Wt[n * 136 + kb2]);
      v8h b0 = bp[0];
      v8h b1 = bp[2];
      v16h B;
#pragma unroll
      for (int j = 0; j < 8; ++j) { B[j] = b0[j]; B[j + 8] = b1[j]; }
      D[tn] = __builtin_amdgcn_wmma_f32_16x16x32_f16(
          false, A[kc], false, B, (short)0, D[tn], false, false);
    }
  }

  // ---- epilogue: bias, optional mish, per-column stats, store f16 in place ----
#pragma unroll
  for (int tn = 0; tn < 8; ++tn) {
    const int col = tn * 16 + (lane & 15);
    const float bb = lbias[col];
    float s = 0.0f, q = 0.0f;
#pragma unroll
    for (int r = 0; r < 8; ++r) {
      float v = D[tn][r] + bb;
      if (do_mish) v = mish_f(v);
      s += v; q += v * v;
      y[(row0 + hh * 8 + r) * kC + col] = (_Float16)v;
    }
    s += __shfl_xor(s, 16, 32);   // combine the two half-wave row groups (same column)
    q += __shfl_xor(q, 16, 32);
    if (hh == 0) { lsum[wave * 128 + col] = s; lsq[wave * 128 + col] = q; }
  }
  __syncthreads();
  if (t < 128) {
    float S = 0.0f, Q = 0.0f;
#pragma unroll
    for (int w = 0; w < 8; ++w) { S += lsum[w * 128 + t]; Q += lsq[w * 128 + t]; }
    const long long pb = (long long)blockIdx.x * 256;
    part[pb + t]       = S;
    part[pb + 128 + t] = Q;
  }
}

// ---------------- final: normalize + per-segment max (segments contiguous, 256 rows) ------------
__global__ void __launch_bounds__(256) k_final(const _Float16* __restrict__ y,
                                               const float* __restrict__ stats,
                                               float* __restrict__ out) {
  __shared__ float l[256];
  const int t = threadIdx.x, c = t & 127, h = t >> 7;
  const long long seg = blockIdx.x;
  const float sc = stats[c], sh = stats[128 + c];
  float m = -3.402823466e+38f;
  const long long base = seg * 256 + (long long)h * 128;
#pragma unroll 4
  for (int r = 0; r < 128; ++r) {
    float v = fmaf((float)y[(base + r) * kC + c], sc, sh);
    m = fmaxf(m, v);
  }
  l[t] = m;
  __syncthreads();
  if (t < 128) out[seg * kC + c] = fmaxf(l[c], l[c + 128]);
}

// ------------------------------------------------------------------------------------------------
extern "C" void kernel_launch(void* const* d_in, const int* in_sizes, int n_in,
                              void* d_out, int out_size, void* d_ws, size_t ws_size,
                              hipStream_t stream) {
  (void)in_sizes; (void)n_in; (void)out_size;
  if (ws_size < WS_NEED) return;

  const float* points  = (const float*)d_in[0];
  // d_in[1] = segment_ids (unused: segments are contiguous equal-length blocks)
  const float* w_first = (const float*)d_in[2];
  const float* b_first = (const float*)d_in[3];
  const float* mid_g   = (const float*)d_in[4];
  const float* mid_be  = (const float*)d_in[5];
  const float* mid_w   = (const float*)d_in[6];
  const float* mid_b   = (const float*)d_in[7];
  const float* last_g  = (const float*)d_in[8];
  const float* last_be = (const float*)d_in[9];
  float* out = (float*)d_out;

  char* ws = (char*)d_ws;
  _Float16* y  = (_Float16*)(ws + Y_OFF);
  float* part  = (float*)(ws + PART_OFF);
  float* part2 = (float*)(ws + PART2_OFF);
  float* stats = (float*)(ws + STATS_OFF);

  k_first<<<kGridMain, 256, 0, stream>>>(points, w_first, b_first, y, part);
  for (int i = 0; i < 4; ++i) {
    k_reduce1<<<256, 256, 0, stream>>>(part, part2);
    k_finalize<<<1, 256, 0, stream>>>(part2, mid_g + i * kC, mid_be + i * kC, stats);
    k_mid<<<kGridMain, 256, 0, stream>>>(y, mid_w + (size_t)i * kC * kC, mid_b + i * kC,
                                         stats, part, (i < 3) ? 1 : 0);
  }
  k_reduce1<<<256, 256, 0, stream>>>(part, part2);
  k_finalize<<<1, 256, 0, stream>>>(part2, last_g, last_be, stats);
  k_final<<<kSegs, 256, 0, stream>>>(y, stats, out);
}
